// GPT2Attention_44779329028789
// MI455X (gfx1250) — matrix operand order repaired and checked by reading, not compiled, not run on hardware
//
#include <hip/hip_runtime.h>
#include <hip/hip_bf16.h>

// ---------------------------------------------------------------------------
// GPT-2 attention block for MI455X (gfx1250, wave32, WMMA).
// f32->bf16 casts (+ one-time weight transpose) -> QKV GEMM (wmma bf16,
// double-buffered async global->LDS A staging, B direct from L2-resident
// transposed weights) -> flash attention (wmma bf16) -> proj GEMM.
// ---------------------------------------------------------------------------

typedef __bf16 bf16_t;
typedef __bf16 v16bf __attribute__((ext_vector_type(16)));
typedef __bf16 v8bf  __attribute__((ext_vector_type(8)));
typedef float  v8f   __attribute__((ext_vector_type(8)));
typedef int    v4i   __attribute__((ext_vector_type(4)));

#define B_DIM   2
#define SEQ     2048
#define HID     2048
#define H3      6144
#define NHEAD   16
#define DHEAD   128

#if __has_builtin(__builtin_amdgcn_global_load_async_to_lds_b128)
#define USE_ASYNC_LDS 1
#else
#define USE_ASYNC_LDS 0
#endif

// b128 async copy uses typed int4 pointers (per clang-22 diag):
#define AS1V4(p) ((__attribute__((address_space(1))) v4i*)(p))
#define AS3V4(p) ((__attribute__((address_space(3))) v4i*)(p))

template <int N>
__device__ __forceinline__ void wait_async() {
#if __has_builtin(__builtin_amdgcn_s_wait_asynccnt)
  __builtin_amdgcn_s_wait_asynccnt(N);
#else
  asm volatile("s_wait_asynccnt %0" :: "i"(N) : "memory");
#endif
}

__device__ __forceinline__ v8f wmma_bf16(v16bf a, v16bf b, v8f c) {
  // D = A(16x32 bf16) x B(32x16 bf16) + C(16x16 f32)
  return __builtin_amdgcn_wmma_f32_16x16x32_bf16(
      /*neg_a=*/false, a, /*neg_b=*/false, b,
      /*c_mod=*/(short)0, c, /*reuse_a=*/false, /*reuse_b=*/false);
}

// ---------------------------------------------------------------------------
// f32 -> bf16 cast (n multiple of 4)
// ---------------------------------------------------------------------------
__global__ void cast_f32_bf16(const float* __restrict__ src,
                              bf16_t* __restrict__ dst, int n) {
  int i = (blockIdx.x * blockDim.x + threadIdx.x) * 4;
  if (i + 3 < n) {
    float4 f = *(const float4*)(src + i);
    dst[i + 0] = (bf16_t)f.x;
    dst[i + 1] = (bf16_t)f.y;
    dst[i + 2] = (bf16_t)f.z;
    dst[i + 3] = (bf16_t)f.w;
  }
}

// ---------------------------------------------------------------------------
// One-time weight cast + transpose: src f32 [K][N] -> dst bf16 [N][K].
// 32x32 tiles through LDS, both sides coalesced. block (32,8).
// ---------------------------------------------------------------------------
__global__ void __launch_bounds__(256)
cast_transpose_bf16(const float* __restrict__ src, bf16_t* __restrict__ dst,
                    int K, int N) {
  __shared__ float t[32][33];
  const int tx = threadIdx.x, ty = threadIdx.y;
  const int nb = blockIdx.x * 32, kb = blockIdx.y * 32;
#pragma unroll
  for (int i = 0; i < 4; ++i)
    t[ty + i * 8][tx] = src[(size_t)(kb + ty + i * 8) * N + nb + tx];
  __syncthreads();
#pragma unroll
  for (int i = 0; i < 4; ++i)
    dst[(size_t)(nb + ty + i * 8) * K + kb + tx] = (bf16_t)t[tx][ty + i * 8];
}

// ---------------------------------------------------------------------------
// Tiled GEMM + bias:  C[M,N] = A[M,K](bf16) * B^T (BT is bf16 [N][K]) + bias.
// Block tile 128x64, K-step 32, 128 threads (4 waves); wave computes 32x64.
// A tile: double-buffered GLOBAL_LOAD_ASYNC_TO_LDS_B128 (DMA overlaps WMMA).
// B fragments: direct 32B global loads from L2-resident transposed weights.
// outF32 != 0 -> f32 output, else bf16 output.
// ---------------------------------------------------------------------------
#define GTM 128
#define GTN 64
#define GTK 32

__global__ void __launch_bounds__(128)
gemm_bias(const bf16_t* __restrict__ A, const bf16_t* __restrict__ BT,
          const float* __restrict__ bias, void* __restrict__ Cout,
          int M, int N, int K, int outF32) {
  __shared__ bf16_t as[2][GTM][40];   // double-buffered A tile (80B rows)

  const int tid  = threadIdx.x;
  const int lane = tid & 31;
  const int wave = tid >> 5;
  const int mb = blockIdx.y * GTM;
  const int nb = blockIdx.x * GTN;

  const int mloc = lane & 15;
  const int k0   = (lane < 16) ? 0 : 16;   // A-fragment K base
  const int kk0  = (lane < 16) ? 0 : 16;   // B-fragment K base
  const int mhi  = (lane < 16) ? 0 : 8;    // C-tile row offset

  const bf16_t* arow = A + (size_t)(mb + tid) * K;   // this thread's A row

  v8f acc[2][4] = {};

  // prologue: start DMA of tile 0 into buffer 0
#if USE_ASYNC_LDS
#pragma unroll
  for (int c = 0; c < 4; ++c)
    __builtin_amdgcn_global_load_async_to_lds_b128(
        AS1V4(arow + c * 8), AS3V4(&as[0][tid][c * 8]), 0, 0);
#else
#pragma unroll
  for (int c = 0; c < 4; ++c)
    *(v8bf*)&as[0][tid][c * 8] = *(const v8bf*)(arow + c * 8);
#endif

  const int nsteps = K / GTK;
  for (int t = 0; t < nsteps; ++t) {
    const int kb  = t * GTK;
    const int cur = t & 1;
    const int nxt = cur ^ 1;
    const bool more = (t + 1) < nsteps;

    // kick off DMA for next tile while we compute this one
    if (more) {
      const bf16_t* srow = arow + kb + GTK;
#if USE_ASYNC_LDS
#pragma unroll
      for (int c = 0; c < 4; ++c)
        __builtin_amdgcn_global_load_async_to_lds_b128(
            AS1V4(srow + c * 8), AS3V4(&as[nxt][tid][c * 8]), 0, 0);
#else
#pragma unroll
      for (int c = 0; c < 4; ++c)
        *(v8bf*)&as[nxt][tid][c * 8] = *(const v8bf*)(srow + c * 8);
#endif
      // warm L2 for the B rows of the next K-step
      __builtin_prefetch(BT + (size_t)(nb + (tid & 63)) * K + kb + GTK, 0, 0);
    }
#if USE_ASYNC_LDS
    // in-order completion: drain only the older tile's 4 DMA ops
    if (more) wait_async<4>();
    else      wait_async<0>();
#endif
    __syncthreads();

    // ---- A fragments for this wave's two 16-row strips ----
    v16bf a[2];
#pragma unroll
    for (int s = 0; s < 2; ++s) {
      const bf16_t* ap = &as[cur][wave * 32 + s * 16 + mloc][0];
      v8bf lo = *(const v8bf*)(ap + k0);
      v8bf hi = *(const v8bf*)(ap + 8 + k0);
#pragma unroll
      for (int e = 0; e < 8; ++e) { a[s][e] = lo[e]; a[s][8 + e] = hi[e]; }
    }
    // ---- B fragments straight from transposed weights (L2 resident) ----
#pragma unroll
    for (int j = 0; j < 4; ++j) {
      v16bf bfrag = *(const v16bf*)(BT + (size_t)(nb + j * 16 + mloc) * K + kb + kk0);
      acc[0][j] = wmma_bf16(a[0], bfrag, acc[0][j]);
      acc[1][j] = wmma_bf16(a[1], bfrag, acc[1][j]);
    }
    __syncthreads();   // protect buffer re-targeted by next iteration's DMA
  }

  // ---- epilogue: bias + store (C layout: VGPR r -> row r (+8 high half)) ----
#pragma unroll
  for (int j = 0; j < 4; ++j) {
    int col = nb + j * 16 + mloc;
    float bv = bias[col];
#pragma unroll
    for (int s = 0; s < 2; ++s) {
#pragma unroll
      for (int r = 0; r < 8; ++r) {
        int row = mb + wave * 32 + s * 16 + r + mhi;
        float v = acc[s][j][r] + bv;
        if (outF32) ((float*)Cout)[(size_t)row * N + col] = v;
        else        ((bf16_t*)Cout)[(size_t)row * N + col] = (bf16_t)v;
      }
    }
  }
}

// ---------------------------------------------------------------------------
// Flash attention (causal). qkv: [B*S, 3H] bf16 (Q|K|V packed).
// One wave = one 16-query tile; 4 waves/block with private LDS slices.
// 32 keys per iteration: 8 WMMA for QK^T, online softmax, 8 WMMA for PV.
// ---------------------------------------------------------------------------
__global__ void __launch_bounds__(128)
flash_attn(const bf16_t* __restrict__ qkv, bf16_t* __restrict__ attn_out) {
  __shared__ bf16_t p_lds[4][16][40];    // probs tile (bf16), per wave
  __shared__ bf16_t v_lds[4][DHEAD][48]; // V transposed [dim][key], per wave

  const int lane = threadIdx.x & 31;
  const int wave = threadIdx.x >> 5;
  const int b = blockIdx.y / NHEAD;
  const int h = blockIdx.y % NHEAD;
  const int qt = blockIdx.x * 4 + wave;
  const int q0 = qt * 16;

  const int mloc = lane & 15;
  const int k0   = (lane < 16) ? 0 : 16;
  const int kk0  = (lane < 16) ? 0 : 16;
  const int mhi  = (lane < 16) ? 0 : 8;

  const size_t base = (size_t)b * SEQ * H3;
  const bf16_t* qbase = qkv + base +            (size_t)h * DHEAD;
  const bf16_t* kbase = qkv + base + HID     +  (size_t)h * DHEAD;
  const bf16_t* vbase = qkv + base + 2 * HID +  (size_t)h * DHEAD;

  // Q fragments (loop-invariant): 4 chunks of K=32 over d=128
  v16bf aq[4];
#pragma unroll
  for (int dc = 0; dc < 4; ++dc) {
    const bf16_t* qrow = qbase + (size_t)(q0 + mloc) * H3 + dc * 32;
    v8bf lo = *(const v8bf*)(qrow + k0);
    v8bf hi = *(const v8bf*)(qrow + 8 + k0);
#pragma unroll
    for (int e = 0; e < 8; ++e) { aq[dc][e] = lo[e]; aq[dc][8 + e] = hi[e]; }
  }

  v8f acc[8] = {};                 // O accumulator: 8 dim-chunks of 16
  float row_max[8], row_sum[8];
#pragma unroll
  for (int r = 0; r < 8; ++r) { row_max[r] = -3.0e38f; row_sum[r] = 0.f; }

  const float scale = 0.08838834764831845f;  // 1/sqrt(128)

  for (int kb = 0; kb <= q0 + 15; kb += 32) {
    // ---- scores: Q(16x128) . K^T -> two 16x16 tiles (keys kb..kb+31) ----
    v8f s0 = {}, s1 = {};
#pragma unroll
    for (int dc = 0; dc < 4; ++dc) {
      const bf16_t* kr0 = kbase + (size_t)(kb + mloc)      * H3 + dc * 32 + kk0;
      const bf16_t* kr1 = kbase + (size_t)(kb + 16 + mloc) * H3 + dc * 32 + kk0;
      v16bf bk0 = *(const v16bf*)kr0;
      v16bf bk1 = *(const v16bf*)kr1;
      s0 = wmma_bf16(aq[dc], bk0, s0);
      s1 = wmma_bf16(aq[dc], bk1, s1);
    }

    // ---- stage V tile transposed: lane <-> key row ----
    const bf16_t* vrow = vbase + (size_t)(kb + lane) * H3;
#pragma unroll
    for (int c = 0; c < 16; ++c) {
      v8bf vv = *(const v8bf*)(vrow + c * 8);
#pragma unroll
      for (int e = 0; e < 8; ++e) v_lds[wave][c * 8 + e][lane] = vv[e];
    }

    // ---- online softmax per owned row (16-lane shuffle reductions) ----
    float corr[8];
#pragma unroll
    for (int r = 0; r < 8; ++r) {
      int m = r + mhi;
      int qidx = q0 + m;
      float x0 = s0[r] * scale;
      float x1 = s1[r] * scale;
      if (kb + mloc      > qidx) x0 = -3.0e38f;   // causal mask
      if (kb + 16 + mloc > qidx) x1 = -3.0e38f;
      float tmax = fmaxf(x0, x1);
#pragma unroll
      for (int off = 1; off < 16; off <<= 1)
        tmax = fmaxf(tmax, __shfl_xor(tmax, off, 32));
      float nmax = fmaxf(row_max[r], tmax);
      float c = __expf(row_max[r] - nmax);
      row_max[r] = nmax;
      float p0 = __expf(x0 - nmax);
      float p1 = __expf(x1 - nmax);
      float ps = p0 + p1;
#pragma unroll
      for (int off = 1; off < 16; off <<= 1)
        ps += __shfl_xor(ps, off, 32);
      row_sum[r] = row_sum[r] * c + ps;
      corr[r] = c;
      p_lds[wave][m][mloc]      = (bf16_t)p0;
      p_lds[wave][m][16 + mloc] = (bf16_t)p1;
    }

    // rescale O accumulators
#pragma unroll
    for (int j = 0; j < 8; ++j)
#pragma unroll
      for (int r = 0; r < 8; ++r) acc[j][r] *= corr[r];

    __builtin_amdgcn_wave_barrier();   // order LDS writes vs reads in-wave

    // ---- P(16x32 bf16) @ V(32x128) ----
    v16bf pa;
    {
      const bf16_t* pp = &p_lds[wave][mloc][0];
      v8bf lo = *(const v8bf*)(pp + k0);
      v8bf hi = *(const v8bf*)(pp + 8 + k0);
#pragma unroll
      for (int e = 0; e < 8; ++e) { pa[e] = lo[e]; pa[8 + e] = hi[e]; }
    }
#pragma unroll
    for (int j = 0; j < 8; ++j) {
      v16bf vb = *(const v16bf*)&v_lds[wave][j * 16 + mloc][kk0];
      acc[j] = wmma_bf16(pa, vb, acc[j]);
    }
    __builtin_amdgcn_wave_barrier();
  }

  // ---- normalize and store attn output (bf16, [B*S, H]) ----
#pragma unroll
  for (int r = 0; r < 8; ++r) {
    int m = r + mhi;
    float inv = 1.0f / row_sum[r];
    size_t row = (size_t)b * SEQ + q0 + m;
#pragma unroll
    for (int j = 0; j < 8; ++j) {
      int col = h * DHEAD + j * 16 + mloc;
      attn_out[row * HID + col] = (bf16_t)(acc[j][r] * inv);
    }
  }
}

// ---------------------------------------------------------------------------
// Launcher
// ---------------------------------------------------------------------------
extern "C" void kernel_launch(void* const* d_in, const int* in_sizes, int n_in,
                              void* d_out, int out_size, void* d_ws, size_t ws_size,
                              hipStream_t stream) {
  (void)in_sizes; (void)n_in; (void)out_size; (void)ws_size;

  const float* hidden = (const float*)d_in[0];   // [2,2048,2048]
  const float* w_attn = (const float*)d_in[1];   // [2048,6144]
  const float* b_attn = (const float*)d_in[2];   // [6144]
  const float* w_proj = (const float*)d_in[3];   // [2048,2048]
  const float* b_proj = (const float*)d_in[4];   // [2048]
  float* out = (float*)d_out;                    // [2,2048,2048] f32

  const size_t M = (size_t)B_DIM * SEQ;          // 4096
  bf16_t* hidden_bf = (bf16_t*)d_ws;
  bf16_t* wattnT_bf = hidden_bf + M * HID;       // [3H][H] transposed
  bf16_t* wprojT_bf = wattnT_bf + (size_t)HID * H3;
  bf16_t* qkv_bf    = wprojT_bf + (size_t)HID * HID;
  bf16_t* attn_bf   = qkv_bf + M * H3;

  // hidden cast
  {
    int n = (int)(M * HID);
    cast_f32_bf16<<<n / 1024, 256, 0, stream>>>(hidden, hidden_bf, n);
  }
  // weight cast + transpose (one-time; result stays hot in 192MB L2)
  {
    dim3 grid(H3 / 32, HID / 32), blk(32, 8);
    cast_transpose_bf16<<<grid, blk, 0, stream>>>(w_attn, wattnT_bf, HID, H3);
  }
  {
    dim3 grid(HID / 32, HID / 32), blk(32, 8);
    cast_transpose_bf16<<<grid, blk, 0, stream>>>(w_proj, wprojT_bf, HID, HID);
  }

  // QKV projection: [4096,2048] @ [2048,6144] + b_attn -> bf16
  {
    dim3 grid(H3 / GTN, (int)(M / GTM));
    gemm_bias<<<grid, 128, 0, stream>>>(hidden_bf, wattnT_bf, b_attn,
                                        (void*)qkv_bf, (int)M, H3, HID, 0);
  }

  // causal flash attention per (b,h), 4 query-tiles (waves) per block
  {
    dim3 grid(SEQ / 64, B_DIM * NHEAD);
    flash_attn<<<grid, 128, 0, stream>>>(qkv_bf, attn_bf);
  }

  // output projection: [4096,2048] @ [2048,2048] + b_proj -> f32 out
  {
    dim3 grid(HID / GTN, (int)(M / GTM));
    gemm_bias<<<grid, 128, 0, stream>>>(attn_bf, wprojT_bf, b_proj,
                                        (void*)out, (int)M, HID, HID, 1);
  }
}
